// PhysicsConstrainedAttention_69140383531369
// MI455X (gfx1250) — matrix operand order repaired and checked by reading, not compiled, not run on hardware
//
#include <hip/hip_runtime.h>
#include <hip/hip_bf16.h>
#include <math.h>

typedef _Float16 half_t;
typedef __attribute__((ext_vector_type(16))) _Float16 v16h;
typedef __attribute__((ext_vector_type(8)))  _Float16 v8h;
typedef __attribute__((ext_vector_type(8)))  float    v8f;

#define BATCH   2
#define CH      128
#define NHEADS  8
#define HD      16
#define HWDIM   64
#define NTOK    4096        // 64*64
#define QKVC    384
#define LN_EPS  1e-5f
#define QSCALE  0.25f       // hd^-0.5 = 16^-0.5

#if defined(__has_builtin)
#if __has_builtin(__builtin_amdgcn_tensor_load_to_lds) && \
    __has_builtin(__builtin_amdgcn_s_wait_tensorcnt)
#define USE_TDM 1
#endif
#endif

static __device__ __forceinline__ v16h make_v16(v8h lo, v8h hi) {
  return __builtin_shufflevector(lo, hi, 0,1,2,3,4,5,6,7,8,9,10,11,12,13,14,15);
}

static __device__ __forceinline__ v8f wmma_f16(v16h a, v16h b, v8f c) {
  // D = A(16x32 f16) * B(32x16 f16) + C(16x16 f32)
  return __builtin_amdgcn_wmma_f32_16x16x32_f16(false, a, false, b, (short)0, c,
                                                false, false);
}

#ifdef USE_TDM
typedef __attribute__((ext_vector_type(4))) unsigned int u32x4;
typedef __attribute__((ext_vector_type(8))) int i32x8;
typedef __attribute__((ext_vector_type(4))) int i32x4;

static __device__ __forceinline__ unsigned lds_off(const void* p) {
  // generic LDS pointer = {shared aperture, offset[31:0]}; low 32 bits = LDS addr
  return (unsigned)(size_t)p;
}

// 1-D TDM tile load: nelem contiguous 2-byte elements -> LDS (D# per ISA ch.8)
static __device__ __forceinline__ void tdm_load_1d(unsigned lds_addr,
                                                   const void* gptr,
                                                   unsigned nelem) {
  unsigned long long ga = (unsigned long long)(size_t)gptr;
  u32x4 g0;
  g0[0] = 1u;                                  // count=1, user descriptor
  g0[1] = lds_addr;                            // LDS byte address
  g0[2] = (unsigned)ga;                        // global_addr[31:0]
  g0[3] = (unsigned)(ga >> 32) | (2u << 30);   // global_addr[56:32] | type=2
  i32x8 g1;
  g1[0] = (int)(1u << 16);                     // wg_mask=0, data_size=1 (2B)
  g1[1] = 0;                                   // tensor_dim0[15:0] (dim0=1<<30)
  g1[2] = (int)(0x4000u | (1u << 16));         // dim0[31:16], tensor_dim1=1
  g1[3] = (int)(nelem << 16);                  // tile_dim0 = nelem
  g1[4] = 0;                                   // tile_dim1=0 (1-D), tile_dim2=0
  g1[5] = (int)nelem;                          // tensor_dim0_stride (unused 1-D)
  g1[6] = 0;
  g1[7] = 0;
  i32x4 gz = {0, 0, 0, 0};
#if __clang_major__ >= 23
  i32x8 gz8 = {0, 0, 0, 0, 0, 0, 0, 0};
  __builtin_amdgcn_tensor_load_to_lds(g0, g1, gz, gz, gz8, 0);
#else
  __builtin_amdgcn_tensor_load_to_lds(g0, g1, gz, gz, 0);
#endif
}
#endif  // USE_TDM

// ---------------------------------------------------------------------------
// 0) one-time weight convert: f32 [K,N] -> f16 n-major [N,K] so B-fragments
//    are two contiguous b128 loads per lane.
// ---------------------------------------------------------------------------
__global__ void __launch_bounds__(256)
convert_weights(const float* __restrict__ wqkv, const float* __restrict__ wproj,
                half_t* __restrict__ wqkv_h, half_t* __restrict__ wproj_h) {
  int idx = blockIdx.x * 256 + threadIdx.x;
  const int nqkv = QKVC * CH;
  if (idx < nqkv) {
    int n = idx / CH, k = idx - n * CH;
    wqkv_h[idx] = (half_t)wqkv[k * QKVC + n];
  } else {
    int j = idx - nqkv;
    if (j < CH * CH) {
      int n = j / CH, k = j - n * CH;
      wproj_h[j] = (half_t)wproj[k * CH + n];
    }
  }
}

// ---------------------------------------------------------------------------
// 1) LayerNorm + QKV GEMM. One block = 16 tokens. 4 waves x 6 n-tiles, K=128.
//    Q pre-scaled by hd^-0.5. Output layout: [B][nh][N][16] f16.
// ---------------------------------------------------------------------------
__global__ void __launch_bounds__(128)
ln_qkv_kernel(const float* __restrict__ x, const half_t* __restrict__ wqkv_h,
              const float* __restrict__ ln_g, const float* __restrict__ ln_b,
              half_t* __restrict__ qh, half_t* __restrict__ kh,
              half_t* __restrict__ vh) {
  __shared__ __align__(16) float  xs[16 * CH];
  __shared__ float mu[16], rs[16];
  __shared__ __align__(16) half_t an[16 * CH];

  const int blk = blockIdx.x;               // 0..511
  const int b  = blk / (NTOK / 16);
  const int n0 = (blk % (NTOK / 16)) * 16;
  const int t  = threadIdx.x;               // == channel, 0..127

  const float* xp = x + ((size_t)b * CH + t) * NTOK + n0;  // [B,C,N] layout
  #pragma unroll
  for (int j = 0; j < 16; ++j) xs[j * CH + t] = xp[j];
  __syncthreads();

  if (t < 16) {
    float s = 0.f, s2 = 0.f;
    for (int c = 0; c < CH; ++c) { float v = xs[t * CH + c]; s += v; s2 += v * v; }
    float m = s * (1.0f / CH);
    float var = s2 * (1.0f / CH) - m * m;
    mu[t] = m;
    rs[t] = rsqrtf(var + LN_EPS);
  }
  __syncthreads();

  const float g = ln_g[t], bb = ln_b[t];
  #pragma unroll
  for (int j = 0; j < 16; ++j)
    an[j * CH + t] = (half_t)((xs[j * CH + t] - mu[j]) * rs[j] * g + bb);
  __syncthreads();

  const int wave = t >> 5, lane = t & 31;
  const int m16 = lane & 15, hb = lane >> 4;

  for (int i = 0; i < 6; ++i) {
    const int nt = wave + 4 * i;            // 24 n-tiles of 16 cols
    v8f acc = {};
    #pragma unroll
    for (int kt = 0; kt < 4; ++kt) {
      v8h alo = *(const v8h*)&an[m16 * CH + kt * 32 + hb * 16];
      v8h ahi = *(const v8h*)&an[m16 * CH + kt * 32 + hb * 16 + 8];
      const half_t* bp = wqkv_h + (size_t)(nt * 16 + m16) * CH + kt * 32 + hb * 16;
      v8h blo = *(const v8h*)bp;
      v8h bhi = *(const v8h*)(bp + 8);
      acc = wmma_f16(make_v16(alo, ahi), make_v16(blo, bhi), acc);
    }
    const int jc   = nt * 16 + m16;         // 0..383 output column
    const int mat  = jc >> 7;               // 0=q 1=k 2=v (uniform per tile)
    const int cc   = jc & 127;
    const int head = cc >> 4, d = cc & 15;
    half_t* dst = (mat == 0) ? qh : (mat == 1) ? kh : vh;
    const float mul = (mat == 0) ? QSCALE : 1.0f;
    const size_t rowbase = (size_t)(b * NHEADS + head) * NTOK + n0 + 8 * hb;
    #pragma unroll
    for (int v = 0; v < 8; ++v)
      dst[(rowbase + v) * HD + d] = (half_t)(acc[v] * mul);
  }
}

// ---------------------------------------------------------------------------
// 2) Flash attention: 128 threads = 4 waves, each wave owns 16 query rows;
//    K/V staged 32 keys/iter via TDM (double-buffered), shared by all waves.
//    Per 32 keys: 2 half-K QK^T WMMAs + 1 full-K P*V WMMA.
// ---------------------------------------------------------------------------
__global__ void __launch_bounds__(128)
attn_kernel(const half_t* __restrict__ qh, const half_t* __restrict__ kh,
            const half_t* __restrict__ vh, half_t* __restrict__ xat) {
  __shared__ __align__(16) half_t Klds[2][32 * 16];   // [key][d]
  __shared__ __align__(16) half_t Vraw[2][32 * 16];   // [key][d]
  __shared__ __align__(16) half_t Vt[2][16 * 32];     // [d][key]
  __shared__ __align__(16) half_t Pt[4][16 * 32];     // per-wave P tile

  const int bhid = blockIdx.x >> 6;   // b*8+h
  const int q64  = blockIdx.x & 63;   // 64-row query super-tile
  const int b    = bhid >> 3, h = bhid & 7;
  const int t    = threadIdx.x;
  const int wave = t >> 5, lane = t & 31;
  const int m16  = lane & 15, hb = lane >> 4;

  const v8h zero8 = {};
  const v16h zero16 = make_v16(zero8, zero8);

  // per-wave Q fragment (K dim = head_dim, upper half of K=32 zero)
  const int qrow0 = q64 * 64 + wave * 16;
  const half_t* qbase = qh + ((size_t)bhid * NTOK + qrow0) * HD;
  v16h a_q = (hb == 0) ? make_v16(*(const v8h*)(qbase + m16 * HD), *(const v8h*)(qbase + m16 * HD + 8)) : zero16;

  float m_i[8], l_i[8];
  #pragma unroll
  for (int v = 0; v < 8; ++v) { m_i[v] = -INFINITY; l_i[v] = 0.f; }
  v8f o = {};

  const half_t* kb0 = kh + (size_t)bhid * NTOK * HD;
  const half_t* vb0 = vh + (size_t)bhid * NTOK * HD;
  half_t* Pw = &Pt[wave][0];

  const int NIT = NTOK / 32;          // 128 iterations of 32 keys

  // prologue: stage buffer 0
#ifdef USE_TDM
  if (wave == 0) {
    tdm_load_1d(lds_off(&Klds[0][0]), kb0, 512);
    tdm_load_1d(lds_off(&Vraw[0][0]), vb0, 512);
  }
#else
  {
    const half_t* src = (t < 64) ? kb0 : vb0;
    half_t* dst = (t < 64) ? &Klds[0][0] : &Vraw[0][0];
    const int i = (t & 63) * 8;
    *(v8h*)&dst[i] = *(const v8h*)&src[i];
  }
#endif

  for (int kt = 0; kt < NIT; ++kt) {
    const int cur = kt & 1, nxt = cur ^ 1;
#ifdef USE_TDM
    if (wave == 0) __builtin_amdgcn_s_wait_tensorcnt(0);
#endif
    __syncthreads();                       // staged tiles visible to all waves

    // transpose Vraw[cur] -> Vt[cur]  (512 elems / 128 threads)
    #pragma unroll
    for (int i = 0; i < 4; ++i) {
      const int l = t + 128 * i;           // l = key*16 + d
      Vt[cur][(l & 15) * 32 + (l >> 4)] = Vraw[cur][l];
    }
    __syncthreads();

    // kick off next tile's DMA (overlaps with compute below)
    if (kt + 1 < NIT) {
      const half_t* kpn = kb0 + (size_t)(kt + 1) * 32 * HD;
      const half_t* vpn = vb0 + (size_t)(kt + 1) * 32 * HD;
#ifdef USE_TDM
      if (wave == 0) {
        tdm_load_1d(lds_off(&Klds[nxt][0]), kpn, 512);
        tdm_load_1d(lds_off(&Vraw[nxt][0]), vpn, 512);
      }
#else
      const half_t* src = (t < 64) ? kpn : vpn;
      half_t* dst = (t < 64) ? &Klds[nxt][0] : &Vraw[nxt][0];
      const int i = (t & 63) * 8;
      *(v8h*)&dst[i] = *(const v8h*)&src[i];
#endif
    }

    // ---- S = Q*K^T for 32 keys (two 16-col tiles) ----
    v8h k0lo = *(const v8h*)&Klds[cur][m16 * 16];
    v8h k0hi = *(const v8h*)&Klds[cur][m16 * 16 + 8];
    v8h k1lo = *(const v8h*)&Klds[cur][(16 + m16) * 16];
    v8h k1hi = *(const v8h*)&Klds[cur][(16 + m16) * 16 + 8];
    v16h b_k0 = (hb == 0) ? make_v16(k0lo, k0hi) : zero16;
    v16h b_k1 = (hb == 0) ? make_v16(k1lo, k1hi) : zero16;
    v8f zf = {};
    v8f s0 = wmma_f16(a_q, b_k0, zf);
    v8f s1 = wmma_f16(a_q, b_k1, zf);

    // ---- online softmax over 32 columns (row = v + 8*hb) ----
    #pragma unroll
    for (int v = 0; v < 8; ++v) {
      float av = s0[v], bv = s1[v];
      float mx = fmaxf(av, bv);
      mx = fmaxf(mx, __shfl_xor(mx, 1));
      mx = fmaxf(mx, __shfl_xor(mx, 2));
      mx = fmaxf(mx, __shfl_xor(mx, 4));
      mx = fmaxf(mx, __shfl_xor(mx, 8));
      float nm   = fmaxf(m_i[v], mx);
      float corr = __expf(m_i[v] - nm);
      float p0   = __expf(av - nm);
      float p1   = __expf(bv - nm);
      float ps = p0 + p1;
      ps += __shfl_xor(ps, 1);
      ps += __shfl_xor(ps, 2);
      ps += __shfl_xor(ps, 4);
      ps += __shfl_xor(ps, 8);
      l_i[v] = l_i[v] * corr + ps;
      m_i[v] = nm;
      o[v]  *= corr;
      Pw[(v + 8 * hb) * 32 + m16]      = (half_t)p0;   // C-layout -> LDS
      Pw[(v + 8 * hb) * 32 + 16 + m16] = (half_t)p1;
    }

    // ---- O += P(16x32) * V(32x16): full K=32 WMMA ----
    v8h plo = *(const v8h*)&Pw[m16 * 32 + hb * 16];
    v8h phi = *(const v8h*)&Pw[m16 * 32 + hb * 16 + 8];
    v8h vlo = *(const v8h*)&Vt[cur][m16 * 32 + hb * 16];
    v8h vhi = *(const v8h*)&Vt[cur][m16 * 32 + hb * 16 + 8];
    o = wmma_f16(make_v16(plo, phi), make_v16(vlo, vhi), o);
  }

  #pragma unroll
  for (int v = 0; v < 8; ++v) {
    const int n = qrow0 + v + 8 * hb;
    const int c = h * HD + m16;
    xat[((size_t)b * NTOK + n) * CH + c] = (half_t)(o[v] / l_i[v]);
  }
}

// ---------------------------------------------------------------------------
// 3) Projection GEMM + bias + residual -> out [B,C,H,W] f32.
// ---------------------------------------------------------------------------
__global__ void __launch_bounds__(256)
proj_kernel(const half_t* __restrict__ xat, const half_t* __restrict__ wproj_h,
            const float* __restrict__ bproj, const float* __restrict__ x,
            float* __restrict__ out) {
  const int blk = blockIdx.x;
  const int b  = blk / (NTOK / 16);
  const int n0 = (blk % (NTOK / 16)) * 16;
  const int t  = threadIdx.x;
  const int wave = t >> 5, lane = t & 31;
  const int m16 = lane & 15, hb = lane >> 4;

  v8f acc = {};
  const half_t* arow = xat + ((size_t)b * NTOK + n0 + m16) * CH;
  #pragma unroll
  for (int kt = 0; kt < 4; ++kt) {
    v8h alo = *(const v8h*)(arow + kt * 32 + hb * 16);
    v8h ahi = *(const v8h*)(arow + kt * 32 + hb * 16 + 8);
    const half_t* bp = wproj_h + (size_t)(wave * 16 + m16) * CH + kt * 32 + hb * 16;
    v8h blo = *(const v8h*)bp;
    v8h bhi = *(const v8h*)(bp + 8);
    acc = wmma_f16(make_v16(alo, ahi), make_v16(blo, bhi), acc);
  }
  const int c = wave * 16 + m16;
  const float bias = bproj[c];
  const size_t base = ((size_t)b * CH + c) * NTOK + n0 + 8 * hb;
  v8f r = *(const v8f*)(x + base);           // residual, [B,C,N] contiguous in n
  v8f res;
  #pragma unroll
  for (int v = 0; v < 8; ++v) res[v] = acc[v] + bias + r[v];
  *(v8f*)(out + base) = res;
}

// ---------------------------------------------------------------------------
// 4) physics reg-loss: single pass sums, then finalize.
// ---------------------------------------------------------------------------
__global__ void zero_kernel(float* sums) {
  if (threadIdx.x < 4) sums[threadIdx.x] = 0.f;
}

__global__ void __launch_bounds__(256)
reduce_kernel(const float* __restrict__ out, float* __restrict__ sums) {
  const size_t M = (size_t)BATCH * CH * NTOK;
  float s1 = 0.f, s2 = 0.f, sdx = 0.f, sdy = 0.f;
  for (size_t idx = (size_t)blockIdx.x * 256 + threadIdx.x; idx < M;
       idx += (size_t)gridDim.x * 256) {
    float v = out[idx];
    s1 += v; s2 += v * v;
    const int n = (int)(idx & (NTOK - 1));
    const int w = n & 63, hh = (n >> 6) & 63;
    if (w  < 63) sdx += fabsf(out[idx + 1]  - v);
    if (hh < 63) sdy += fabsf(out[idx + 64] - v);
  }
  #pragma unroll
  for (int m = 16; m > 0; m >>= 1) {
    s1  += __shfl_down(s1, m);
    s2  += __shfl_down(s2, m);
    sdx += __shfl_down(sdx, m);
    sdy += __shfl_down(sdy, m);
  }
  if ((threadIdx.x & 31) == 0) {
    atomicAdd(&sums[0], s1);
    atomicAdd(&sums[1], s2);
    atomicAdd(&sums[2], sdx);
    atomicAdd(&sums[3], sdy);
  }
}

__global__ void finalize_kernel(const float* __restrict__ sums,
                                float* __restrict__ out_scalar) {
  const float M   = (float)((size_t)BATCH * CH * NTOK);
  const float ndx = (float)(BATCH * CH * HWDIM * 63);
  const float smooth = sums[2] / ndx + sums[3] / ndx;
  const float mean   = sums[0] / M;
  const float inten  = sums[1] / M - mean * mean;
  out_scalar[0] = 0.2f * 0.1f * (smooth + 0.1f * inten);
}

// ---------------------------------------------------------------------------
extern "C" void kernel_launch(void* const* d_in, const int* in_sizes, int n_in,
                              void* d_out, int out_size, void* d_ws, size_t ws_size,
                              hipStream_t stream) {
  const float* x     = (const float*)d_in[0];
  const float* wqkv  = (const float*)d_in[1];
  const float* wproj = (const float*)d_in[2];
  const float* bproj = (const float*)d_in[3];
  const float* lng   = (const float*)d_in[4];
  const float* lnb   = (const float*)d_in[5];
  float* out = (float*)d_out;

  char* ws = (char*)d_ws;
  half_t* wqkv_h  = (half_t*)ws; ws += (size_t)QKVC * CH * sizeof(half_t);
  half_t* wproj_h = (half_t*)ws; ws += (size_t)CH * CH * sizeof(half_t);
  const size_t qkv_elems = (size_t)BATCH * NHEADS * NTOK * HD;
  half_t* qh = (half_t*)ws; ws += qkv_elems * sizeof(half_t);
  half_t* kh = (half_t*)ws; ws += qkv_elems * sizeof(half_t);
  half_t* vh = (half_t*)ws; ws += qkv_elems * sizeof(half_t);
  half_t* xat = (half_t*)ws; ws += (size_t)BATCH * NTOK * CH * sizeof(half_t);
  float* sums = (float*)ws;

  convert_weights<<<(QKVC * CH + CH * CH + 255) / 256, 256, 0, stream>>>(
      wqkv, wproj, wqkv_h, wproj_h);
  ln_qkv_kernel<<<BATCH * (NTOK / 16), 128, 0, stream>>>(
      x, wqkv_h, lng, lnb, qh, kh, vh);
  attn_kernel<<<BATCH * NHEADS * (NTOK / 64), 128, 0, stream>>>(qh, kh, vh, xat);
  proj_kernel<<<BATCH * (NTOK / 16), 256, 0, stream>>>(xat, wproj_h, bproj, x, out);
  zero_kernel<<<1, 32, 0, stream>>>(sums);
  reduce_kernel<<<512, 256, 0, stream>>>(out, sums);
  finalize_kernel<<<1, 1, 0, stream>>>(sums, out + (size_t)BATCH * CH * NTOK);
}
